// SPFSModel_23587960390123
// MI455X (gfx1250) — compile-verified, matrix operand order checked by reading, not run on hardware
//
#include <hip/hip_runtime.h>
#include <hip/hip_bf16.h>
#include <math.h>

// ---------------- model constants ----------------
#define T_ 24
#define D_ 64
#define M_ 16
#define N_ 2048

typedef __attribute__((ext_vector_type(16))) _Float16 v16h;
typedef __attribute__((ext_vector_type(8)))  _Float16 v8h;
typedef __attribute__((ext_vector_type(8)))  float    v8f;

// ---------------- WMMA helpers (CDNA5 16x16x32 f16 -> f32) ----------------

static __device__ __forceinline__ int frag_k(int e, int hi) {
    // 16-bit A/B fragment K index for element e (0..15), lane-half hi (0/1)
    int v = e >> 1, h = e & 1;
    return ((v >> 2) << 4) + (hi << 3) + ((v & 3) << 1) + h;
}

static __device__ __forceinline__ v8f wmma32(v16h a, v16h b, v8f c) {
    return __builtin_amdgcn_wmma_f32_16x16x32_f16(
        /*neg_a=*/false, a, /*neg_b=*/false, b,
        /*c_mod=*/(short)0, c, /*reuse_a=*/false, /*reuse_b=*/false);
}

// A fragment: rows row0..row0+15 of an f16 row-major matrix (ld elems/row).
// frag_k is contiguous over e=0..7 and e=8..15, so this lowers to 2x b128.
static __device__ __forceinline__ v16h a_frag(const _Float16* buf, int row0, int ld,
                                              int k0, int lane) {
    int m  = row0 + (lane & 15);
    int hi = lane >> 4;
    const _Float16* r = buf + (size_t)m * ld + k0;
    v16h a;
#pragma unroll
    for (int e = 0; e < 16; ++e) a[e] = r[frag_k(e, hi)];
    return a;
}

// A fragment where row m is node (n0+m) at time t inside an (N,T,64) f16 tensor
static __device__ __forceinline__ v16h a_frag_nt(const _Float16* g, int n0, int t,
                                                 int k0, int lane) {
    int m  = lane & 15;
    int hi = lane >> 4;
    const _Float16* r = g + (((size_t)(n0 + m)) * T_ + t) * D_ + k0;
    v16h a;
#pragma unroll
    for (int e = 0; e < 16; ++e) a[e] = r[frag_k(e, hi)];
    return a;
}

// A fragment with all 16 rows identical (broadcast row, e.g. te[t])
static __device__ __forceinline__ v16h a_frag_bcast(const _Float16* row, int k0, int lane) {
    int hi = lane >> 4;
    v16h a;
#pragma unroll
    for (int e = 0; e < 16; ++e) a[e] = row[k0 + frag_k(e, hi)];
    return a;
}

// B fragment from pre-packed fragment-major f16 weights:
// packed[((ct*nkt + kt)*32 + lane)*16 + e]  -> single 32B contiguous load/lane
static __device__ __forceinline__ v16h b_frag_p(const _Float16* packed, int nkt,
                                                int ct, int kt, int lane) {
    return *(const v16h*)(packed + (((size_t)ct * nkt + kt) * 32 + lane) * 16);
}

// Y = act(X @ W^T + b), X: rows x 64 f16 (LDS), Y: rows x 64 f16. 8 waves.
// A fragments hoisted per row-tile; col-tiles interleaved in pairs so
// independent WMMAs absorb the WMMA->VALU hazard window.
static __device__ __forceinline__ void fc_rows(const _Float16* X, _Float16* Y,
                                               const _Float16* Wp, const float* B,
                                               int rows, int act, int tid) {
    int lane = tid & 31, wid = tid >> 5;
    int nrt = rows >> 4;
    int n_lo = lane & 15, hi = lane >> 4;
    for (int rt = wid; rt < nrt; rt += 8) {
        v16h a0 = a_frag(X, rt * 16, D_, 0,  lane);
        v16h a1 = a_frag(X, rt * 16, D_, 32, lane);
#pragma unroll
        for (int cp = 0; cp < 2; ++cp) {
            int ct0 = cp * 2, ct1 = ct0 + 1;
            v8f acc0 = {}, acc1 = {};
            acc0 = wmma32(a0, b_frag_p(Wp, 2, ct0, 0, lane), acc0);
            acc1 = wmma32(a0, b_frag_p(Wp, 2, ct1, 0, lane), acc1);
            acc0 = wmma32(a1, b_frag_p(Wp, 2, ct0, 1, lane), acc0);
            acc1 = wmma32(a1, b_frag_p(Wp, 2, ct1, 1, lane), acc1);
            int nc0 = ct0 * 16 + n_lo, nc1 = ct1 * 16 + n_lo;
            float b0v = B[nc0], b1v = B[nc1];
#pragma unroll
            for (int i = 0; i < 8; ++i) {
                int m = rt * 16 + i + hi * 8;
                float v0 = acc0[i] + b0v, v1 = acc1[i] + b1v;
                if (act == 1) { v0 = fmaxf(v0, 0.0f); v1 = fmaxf(v1, 0.0f); }
                Y[m * D_ + nc0] = (_Float16)v0;
                Y[m * D_ + nc1] = (_Float16)v1;
            }
        }
    }
}

// 16-row FC; 4 waves (one col-tile each). act: 0 none, 1 relu, 2 tanh
static __device__ __forceinline__ void fc16(const _Float16* X, _Float16* Y,
                                            const _Float16* Wp, const float* B,
                                            int act, int tid) {
    int lane = tid & 31, wid = tid >> 5;
    if (wid < 4) {
        int ct = wid;
        v8f acc = {};
        acc = wmma32(a_frag(X, 0, D_, 0,  lane), b_frag_p(Wp, 2, ct, 0, lane), acc);
        acc = wmma32(a_frag(X, 0, D_, 32, lane), b_frag_p(Wp, 2, ct, 1, lane), acc);
        int n = ct * 16 + (lane & 15), hi = lane >> 4;
        float bias = B[n];
#pragma unroll
        for (int i = 0; i < 8; ++i) {
            int m = i + hi * 8;
            float v = acc[i] + bias;
            if (act == 1) v = fmaxf(v, 0.0f);
            else if (act == 2) v = tanhf(v);
            Y[m * D_ + n] = (_Float16)v;
        }
    }
}

// ---------------- param structs ----------------

struct BranchParams {
    const float *x, *w;
    const float *in_W0, *in_b0;
    const _Float16 *in_W1p; const float *in_b1;
    const _Float16 *xp_W0p; const float *xp_b0;
    const _Float16 *xp_W1p; const float *xp_b1;
    const _Float16 *yp_W0p; const float *yp_b0;
    const _Float16 *yp_W1p; const float *yp_b1;
    const _Float16 *dl_W0p; const float *dl_b0;
    const _Float16 *dl_W1p; const float *dl_b1;
    const _Float16 *rs_Wp;  const float *rs_b;
    const _Float16 *x2_W0p; const float *x2_b0;
    const _Float16 *x2_W1p; const float *x2_b1;
    const _Float16 *y2_W0p; const float *y2_b0;
    const _Float16 *y2_W1p; const float *y2_b1;
    const _Float16 *g1_W0p; const float *g1_b0;
    const _Float16 *g1_W1p; const float *g1_b1;
    _Float16 *xg2_out, *u_out;
};

struct GruParams {
    const _Float16 *g2_Wp; const float *g2_b;
    const _Float16 *W_ihp; const float *b_ih;
    const _Float16 *W_hhp; const float *b_hh;
    const _Float16 *xg2, *u;
    _Float16* gi;    // (N,T,192) f16 precomputed input-proj gates
    float* pred;
};

struct HeadParams {
    const float *pred_gp, *pred_fs;
    const _Float16 *a1_Wp; const float *a1_b; const float *a2_W;
    const _Float16 *o0_Wp; const float *o0_b;
    const _Float16 *o1_Wp; const float *o1_b;
    const float *o2_W, *o2_b;
    float* out;
};

// ---------------- kernel 0: weight pre-pack (f32 -> f16 fragment-major) ----

__global__ __launch_bounds__(256) void k_pack(const float* W, _Float16* dst,
                                              int out_dim, int in_dim) {
    int nkt = in_dim >> 5;
    int total = out_dim * in_dim;   // == (out/16)*nkt*32*16
    for (int i = blockIdx.x * 256 + threadIdx.x; i < total; i += gridDim.x * 256) {
        int e    = i & 15;
        int lane = (i >> 4) & 31;
        int kt   = (i >> 9) % nkt;
        int ct   = (i >> 9) / nkt;
        int n = ct * 16 + (lane & 15);
        int k = kt * 32 + frag_k(e, lane >> 4);
        dst[i] = (_Float16)W[(size_t)n * in_dim + k];
    }
}

// ---------------- kernel 2: te_embed (tiny, VALU) ----------------

__global__ __launch_bounds__(256) void k_te(const int* TE,
                                            const float* W0, const float* b0,
                                            const float* W1, const float* b1,
                                            _Float16* te_out) {
    __shared__ float l0[T_ * D_];
    int tid = threadIdx.x;
    for (int i = tid; i < T_ * D_; i += 256) {
        int t = i >> 6, c = i & 63;
        int idx = TE[t];  // one-hot(T) @ W0^T selects column idx of W0 (64 x 24)
        l0[i] = fmaxf(W0[c * T_ + idx] + b0[c], 0.0f);
    }
    __syncthreads();
    for (int i = tid; i < T_ * D_; i += 256) {
        int t = i >> 6, c = i & 63;
        float s = b1[c];
        for (int k = 0; k < D_; ++k) s += W1[c * D_ + k] * l0[t * D_ + k];
        te_out[i] = (_Float16)s;
    }
}

// ---------------- kernel 1: fused branch pipeline ----------------
// grid (N/16, T, 2 branches), block 256 (8 waves). Dynamic LDS (~74 KB).

__global__ __launch_bounds__(256) void k_branch(BranchParams Pgp, BranchParams Pfs) {
    const BranchParams P = (blockIdx.z == 0) ? Pgp : Pfs;
    extern __shared__ char smem_raw[];
    _Float16* big0 = (_Float16*)smem_raw;        // 256x64 f16
    _Float16* big1 = big0 + 256 * D_;            // 256x64 f16
    _Float16* sm0  = big1 + 256 * D_;            // 16x64 f16 x4
    _Float16* sm1  = sm0 + 16 * D_;
    _Float16* sm2  = sm1 + 16 * D_;
    _Float16* sm3  = sm2 + 16 * D_;
    float*    swt  = (float*)(sm3 + 16 * D_);    // 16x16 f32
    float*    w0b  = swt + 256;                  // 128 f32 (in_W0 | in_b0)

    int tid = threadIdx.x;
    int n0  = blockIdx.x * 16;
    int t   = blockIdx.y;

    // warm L2 for the packed weights used first
    __builtin_prefetch(P.in_W1p, 0, 1);
    __builtin_prefetch(P.xp_W0p, 0, 1);

    // stage input-layer weights through LDS (one load instead of 256 streams)
    if (tid < 64) {
        w0b[tid]      = P.in_W0[tid];
        w0b[64 + tid] = P.in_b0[tid];
    }
    // --- load x, aggregation weights ---
    float xv;
    {
        int row = tid;                 // row = node_local*16 + neigh
        int node = row >> 4, m = row & 15;
        xv = P.x[((size_t)(n0 + node) * T_ + t) * M_ + m];
        swt[row] = P.w[(size_t)(n0 + node) * M_ + m];
    }
    __syncthreads();
    // input layer0 (1 -> 64), relu
    for (int c8 = 0; c8 < 8; ++c8) {
        v8h o;
#pragma unroll
        for (int j = 0; j < 8; ++j) {
            int c = c8 * 8 + j;
            o[j] = (_Float16)fmaxf(xv * w0b[c] + w0b[64 + c], 0.0f);
        }
        *(v8h*)(big0 + tid * D_ + c8 * 8) = o;
    }
    __syncthreads();
    // input layer1 (linear): big1 = X1
    fc_rows(big0, big1, P.in_W1p, P.in_b1, 256, 0, tid);
    __syncthreads();
    // neighbor aggregation: yg -> sm0 (16x64); 128 threads x (node, 8-col strip)
    if (tid < 128) {
        int node = tid >> 3, cg = (tid & 7) * 8;
        float acc[8] = {0, 0, 0, 0, 0, 0, 0, 0};
        for (int m = 0; m < 16; ++m) {
            float wv = swt[node * 16 + m];
            v8h xr = *(const v8h*)(big1 + (node * 16 + m) * D_ + cg);
#pragma unroll
            for (int j = 0; j < 8; ++j) acc[j] += wv * (float)xr[j];
        }
        v8h o;
#pragma unroll
        for (int j = 0; j < 8; ++j) o[j] = (_Float16)acc[j];
        *(v8h*)(sm0 + node * D_ + cg) = o;
    }
    __syncthreads();
    // x_post: relu, linear  (big1 -> big0 -> big1)
    fc_rows(big1, big0, P.xp_W0p, P.xp_b0, 256, 1, tid);
    __syncthreads();
    fc_rows(big0, big1, P.xp_W1p, P.xp_b1, 256, 0, tid);   // big1 = xpost
    __syncthreads();
    // y_post: sm0 -> sm1 (relu) -> sm2
    fc16(sm0, sm1, P.yp_W0p, P.yp_b0, 1, tid);
    __syncthreads();
    fc16(sm1, sm2, P.yp_W1p, P.yp_b1, 0, tid);             // sm2 = ypost
    __syncthreads();
    // |ypost - xpost| aggregated over neighbors -> sm0
    if (tid < 128) {
        int node = tid >> 3, cg = (tid & 7) * 8;
        v8h yr = *(const v8h*)(sm2 + node * D_ + cg);
        float acc[8] = {0, 0, 0, 0, 0, 0, 0, 0};
        for (int m = 0; m < 16; ++m) {
            float wv = swt[node * 16 + m];
            v8h xr = *(const v8h*)(big1 + (node * 16 + m) * D_ + cg);
#pragma unroll
            for (int j = 0; j < 8; ++j)
                acc[j] += wv * fabsf((float)yr[j] - (float)xr[j]);
        }
        v8h o;
#pragma unroll
        for (int j = 0; j < 8; ++j) o[j] = (_Float16)acc[j];
        *(v8h*)(sm0 + node * D_ + cg) = o;
    }
    __syncthreads();
    // delta: sm0 -> sm1 (relu) -> sm3 (tanh)
    fc16(sm0, sm1, P.dl_W0p, P.dl_b0, 1, tid);
    __syncthreads();
    fc16(sm1, sm3, P.dl_W1p, P.dl_b1, 2, tid);             // sm3 = delta_out
    __syncthreads();
    // residual: sm0 = delta_out + res(ypost)
    {
        int lane = tid & 31, wid = tid >> 5;
        if (wid < 4) {
            int ct = wid;
            v8f acc = {};
            acc = wmma32(a_frag(sm2, 0, D_, 0,  lane), b_frag_p(P.rs_Wp, 2, ct, 0, lane), acc);
            acc = wmma32(a_frag(sm2, 0, D_, 32, lane), b_frag_p(P.rs_Wp, 2, ct, 1, lane), acc);
            int n = ct * 16 + (lane & 15), hi = lane >> 4;
            float bias = P.rs_b[n];
#pragma unroll
            for (int i = 0; i < 8; ++i) {
                int m = i + hi * 8;
                sm0[m * D_ + n] =
                    (_Float16)(acc[i] + bias + (float)sm3[m * D_ + n]);
            }
        }
    }
    __syncthreads();
    // x2: sm3 -> sm1 (relu) -> sm2 (= xg2), also streamed to global f16
    fc16(sm3, sm1, P.x2_W0p, P.x2_b0, 1, tid);
    __syncthreads();
    {
        int lane = tid & 31, wid = tid >> 5;
        if (wid < 4) {
            int ct = wid;
            v8f acc = {};
            acc = wmma32(a_frag(sm1, 0, D_, 0,  lane), b_frag_p(P.x2_W1p, 2, ct, 0, lane), acc);
            acc = wmma32(a_frag(sm1, 0, D_, 32, lane), b_frag_p(P.x2_W1p, 2, ct, 1, lane), acc);
            int n = ct * 16 + (lane & 15), hi = lane >> 4;
            float bias = P.x2_b1[n];
#pragma unroll
            for (int i = 0; i < 8; ++i) {
                int m = i + hi * 8;
                float v = acc[i] + bias;
                sm2[m * D_ + n] = (_Float16)v;
                P.xg2_out[(((size_t)(n0 + m)) * T_ + t) * D_ + n] = (_Float16)v;
            }
        }
    }
    __syncthreads();
    // y2: sm0 (yg_res) -> sm1 (relu) -> sm3 (= yg2)
    fc16(sm0, sm1, P.y2_W0p, P.y2_b0, 1, tid);
    __syncthreads();
    fc16(sm1, sm3, P.y2_W1p, P.y2_b1, 0, tid);
    __syncthreads();
    // g1: sm2 (xg2) -> sm0 (relu) -> u = exp(-relu(.)) * yg2 -> global f16
    fc16(sm2, sm0, P.g1_W0p, P.g1_b0, 1, tid);
    __syncthreads();
    {
        int lane = tid & 31, wid = tid >> 5;
        if (wid < 4) {
            int ct = wid;
            v8f acc = {};
            acc = wmma32(a_frag(sm0, 0, D_, 0,  lane), b_frag_p(P.g1_W1p, 2, ct, 0, lane), acc);
            acc = wmma32(a_frag(sm0, 0, D_, 32, lane), b_frag_p(P.g1_W1p, 2, ct, 1, lane), acc);
            int n = ct * 16 + (lane & 15), hi = lane >> 4;
            float bias = P.g1_b1[n];
#pragma unroll
            for (int i = 0; i < 8; ++i) {
                int m = i + hi * 8;
                float g = expf(-fmaxf(acc[i] + bias, 0.0f));
                float u = g * (float)sm3[m * D_ + n];
                P.u_out[(((size_t)(n0 + m)) * T_ + t) * D_ + n] = (_Float16)u;
            }
        }
    }
}

// ---------------- kernel 3a: input-projection gates (parallel over t) -----
// gi[n,t,:] = [u(n,t), te(t)] @ W_ih^T + b_ih   (state-independent!)
// grid (N/16, T, 2 branches), block 128 (4 waves). A fragments hoisted.

__global__ __launch_bounds__(128) void k_gi(GruParams Pgp, GruParams Pfs,
                                            const _Float16* te) {
    const GruParams P = (blockIdx.z == 0) ? Pgp : Pfs;
    int tid = threadIdx.x, lane = tid & 31, wid = tid >> 5;
    int n0 = blockIdx.x * 16;
    int t  = blockIdx.y;
    __builtin_prefetch(P.W_ihp, 0, 1);
    v16h au0 = a_frag_nt(P.u, n0, t, 0,  lane);
    v16h au1 = a_frag_nt(P.u, n0, t, 32, lane);
    v16h at0 = a_frag_bcast(te + (size_t)t * D_, 0,  lane);
    v16h at1 = a_frag_bcast(te + (size_t)t * D_, 32, lane);
    for (int ct = wid; ct < 12; ct += 4) {
        v8f acc = {};
        acc = wmma32(au0, b_frag_p(P.W_ihp, 4, ct, 0, lane), acc);
        acc = wmma32(au1, b_frag_p(P.W_ihp, 4, ct, 1, lane), acc);
        acc = wmma32(at0, b_frag_p(P.W_ihp, 4, ct, 2, lane), acc);
        acc = wmma32(at1, b_frag_p(P.W_ihp, 4, ct, 3, lane), acc);
        int n = ct * 16 + (lane & 15), hi = lane >> 4;
        float bi = P.b_ih[n];
#pragma unroll
        for (int i = 0; i < 8; ++i) {
            int m = i + hi * 8;
            P.gi[(((size_t)(n0 + m)) * T_ + t) * 192 + n] = (_Float16)(acc[i] + bi);
        }
    }
}

// ---------------- kernel 3b: recurrent scan (g2 gate + hidden proj) -------
// grid (N/16, 1, 2 branches), block 128 (4 waves), sequential over T.
// B fragments for g2_W and W_hh (and biases) live in registers across the scan.

__global__ __launch_bounds__(128) void k_gru(GruParams Pgp, GruParams Pfs) {
    const GruParams P = (blockIdx.z == 0) ? Pgp : Pfs;
    __shared__ float    sbuf[16 * D_];       // state s (f32)
    __shared__ float    hgf[16 * D_];        // gated state hg (f32)
    __shared__ _Float16 hgh[16 * D_];        // gated state (f16, WMMA A)
    __shared__ float    ghb[16 * 192];       // hidden-proj gates
    __shared__ float    g2b[16 * D_];        // g2 gate
    int tid = threadIdx.x, lane = tid & 31, wid = tid >> 5;
    int n0 = blockIdx.x * 16;

    // loop-invariant operand fragments + biases in registers
    v16h g2f0 = b_frag_p(P.g2_Wp, 2, wid, 0, lane);
    v16h g2f1 = b_frag_p(P.g2_Wp, 2, wid, 1, lane);
    float g2bias = P.g2_b[wid * 16 + (lane & 15)];
    v16h whh[3][2];
    float bh[3];
#pragma unroll
    for (int j = 0; j < 3; ++j) {
        int ct = wid + 4 * j;
        whh[j][0] = b_frag_p(P.W_hhp, 2, ct, 0, lane);
        whh[j][1] = b_frag_p(P.W_hhp, 2, ct, 1, lane);
        bh[j] = P.b_hh[ct * 16 + (lane & 15)];
    }

    for (int i = tid; i < 16 * D_; i += 128) sbuf[i] = 0.0f;
    __syncthreads();

    for (int t = 0; t < T_; ++t) {
        // g2 = exp(-relu(W xg2[t-1] + b)); zero input row for t == 0
        {
            v8f acc = {};
            if (t > 0) {
                acc = wmma32(a_frag_nt(P.xg2, n0, t - 1, 0,  lane), g2f0, acc);
                acc = wmma32(a_frag_nt(P.xg2, n0, t - 1, 32, lane), g2f1, acc);
            }
            int n = wid * 16 + (lane & 15), hi = lane >> 4;
#pragma unroll
            for (int i = 0; i < 8; ++i)
                g2b[(i + hi * 8) * D_ + n] = expf(-fmaxf(acc[i] + g2bias, 0.0f));
        }
        __syncthreads();
        // hg = g2 * s   (8-wide strips, vector LDS stores)
        {
            int m = tid >> 3, cg = (tid & 7) * 8;
            v8f hv;
            v8h hh;
#pragma unroll
            for (int j = 0; j < 8; ++j) {
                float h = g2b[m * D_ + cg + j] * sbuf[m * D_ + cg + j];
                hv[j] = h;
                hh[j] = (_Float16)h;
            }
            *(v8f*)(hgf + m * D_ + cg) = hv;
            *(v8h*)(hgh + m * D_ + cg) = hh;
        }
        __syncthreads();
        // gh = hg @ W_hh^T + b_hh  (192 outs, 12 col tiles over 4 waves)
        {
            v16h a0 = a_frag(hgh, 0, D_, 0,  lane);
            v16h a1 = a_frag(hgh, 0, D_, 32, lane);
            v8f aH0 = {}, aH1 = {}, aH2 = {};
            aH0 = wmma32(a0, whh[0][0], aH0);
            aH1 = wmma32(a0, whh[1][0], aH1);
            aH2 = wmma32(a0, whh[2][0], aH2);
            aH0 = wmma32(a1, whh[0][1], aH0);
            aH1 = wmma32(a1, whh[1][1], aH1);
            aH2 = wmma32(a1, whh[2][1], aH2);
            int n_lo = lane & 15, hi = lane >> 4;
#pragma unroll
            for (int i = 0; i < 8; ++i) {
                int m = i + hi * 8;
                ghb[m * 192 + (wid +  0) * 16 + n_lo] = aH0[i] + bh[0];
                ghb[m * 192 + (wid +  4) * 16 + n_lo] = aH1[i] + bh[1];
                ghb[m * 192 + (wid +  8) * 16 + n_lo] = aH2[i] + bh[2];
            }
        }
        __syncthreads();
        // GRU gate math + state update + vectorized prediction store
        {
            int m = tid >> 3, cg = (tid & 7) * 8;
            const _Float16* gi = P.gi + (((size_t)(n0 + m)) * T_ + t) * 192;
            v8h gir = *(const v8h*)(gi + cg);
            v8h giz = *(const v8h*)(gi + 64 + cg);
            v8h gin = *(const v8h*)(gi + 128 + cg);
            v8f sv;
#pragma unroll
            for (int j = 0; j < 8; ++j) {
                int c = cg + j;
                float hr = ghb[m * 192 + c], hz = ghb[m * 192 + 64 + c],
                      hn = ghb[m * 192 + 128 + c];
                float r  = 1.0f / (1.0f + expf(-((float)gir[j] + hr)));
                float z  = 1.0f / (1.0f + expf(-((float)giz[j] + hz)));
                float nn = tanhf((float)gin[j] + r * hn);
                float h  = hgf[m * D_ + c];
                float s  = (1.0f - z) * nn + z * h;
                sbuf[m * D_ + c] = s;
                sv[j] = s;
            }
            *(v8f*)(P.pred + (((size_t)(n0 + m)) * T_ + t) * D_ + cg) = sv;
        }
        __syncthreads();
    }
}

// ---------------- kernel 4: attention fusion + output head ----------------
// grid (N*T/16), block 128 (4 waves); each block handles 16 (n,t) rows.

__global__ __launch_bounds__(128) void k_head(HeadParams P) {
    __shared__ _Float16 pb[32 * D_];
    __shared__ float    pf32[32 * D_];
    __shared__ _Float16 ab[32 * D_];
    __shared__ float    score[32];
    __shared__ _Float16 fb[16 * D_];
    __shared__ _Float16 h0[16 * D_];
    __shared__ _Float16 h1[16 * D_];
    int tid = threadIdx.x, lane = tid & 31, wid = tid >> 5;
    int q0 = blockIdx.x * 16;  // q = n*T + t (pred is contiguous in q)

    // vectorized pred load: 256 (row, 8-col) strips over 128 threads
    for (int u = tid; u < 256; u += 128) {
        int row = u >> 3, cg = (u & 7) * 8;
        int m = row >> 1, br = row & 1;
        const float* src = (br ? P.pred_fs : P.pred_gp) +
                           (size_t)(q0 + m) * D_ + cg;
        v8f vv = *(const v8f*)src;
        v8h hh;
#pragma unroll
        for (int j = 0; j < 8; ++j) hh[j] = (_Float16)vv[j];
        *(v8f*)(pf32 + row * D_ + cg) = vv;
        *(v8h*)(pb + row * D_ + cg)   = hh;
    }
    __syncthreads();
    // attn_fc1 (tanh): 32x64, 2 row tiles x 4 col tiles over 4 waves
    for (int tix = wid; tix < 8; tix += 4) {
        int rt = tix >> 2, ct = tix & 3;
        v8f acc = {};
        acc = wmma32(a_frag(pb, rt * 16, D_, 0,  lane), b_frag_p(P.a1_Wp, 2, ct, 0, lane), acc);
        acc = wmma32(a_frag(pb, rt * 16, D_, 32, lane), b_frag_p(P.a1_Wp, 2, ct, 1, lane), acc);
        int n = ct * 16 + (lane & 15), hi = lane >> 4;
        float bias = P.a1_b[n];
#pragma unroll
        for (int i = 0; i < 8; ++i) {
            int m = rt * 16 + i + hi * 8;
            ab[m * D_ + n] = (_Float16)tanhf(acc[i] + bias);
        }
    }
    __syncthreads();
    // attn_fc2 (64 -> 1, no bias): one score per pred row
    if (tid < 32) {
        float s = 0.0f;
        for (int c = 0; c < D_; ++c) s += P.a2_W[c] * (float)ab[tid * D_ + c];
        score[tid] = s;
    }
    __syncthreads();
    // 2-way softmax + fuse (8-wide strips)
    {
        int m = tid >> 3, cg = (tid & 7) * 8;
        float s0 = score[2 * m], s1 = score[2 * m + 1];
        float mx = fmaxf(s0, s1);
        float e0 = expf(s0 - mx), e1 = expf(s1 - mx);
        float inv = 1.0f / (e0 + e1);
        float w0 = e0 * inv, w1 = e1 * inv;
        v8f p0 = *(const v8f*)(pf32 + (2 * m) * D_ + cg);
        v8f p1 = *(const v8f*)(pf32 + (2 * m + 1) * D_ + cg);
        v8h hh;
#pragma unroll
        for (int j = 0; j < 8; ++j) hh[j] = (_Float16)(w0 * p0[j] + w1 * p1[j]);
        *(v8h*)(fb + m * D_ + cg) = hh;
    }
    __syncthreads();
    fc16(fb, h0, P.o0_Wp, P.o0_b, 1, tid);
    __syncthreads();
    fc16(h0, h1, P.o1_Wp, P.o1_b, 1, tid);
    __syncthreads();
    if (tid < 16) {
        float s = P.o2_b[0];
        for (int c = 0; c < D_; ++c) s += P.o2_W[c] * (float)h1[tid * D_ + c];
        P.out[q0 + tid] = s;  // *STD + MEAN is identity (STD=1, MEAN=0)
    }
}

// ---------------- launch ----------------
// Input order (jax tree-flatten, dict keys sorted):
//  0 x_gp  1 x_fs  2 gp  3 fs  4 TE
//  5,6 attn_fc1 W,b | 7 attn_fc2 W
//  8-11 cell_fs (W_hh,W_ih,b_hh,b_ih) | 12-15 cell_gp
//  16-19 delta_fs (W0,b0,W1,b1) | 20-23 delta_gp
//  24-27 g1_fs | 28-31 g1_gp | 32,33 g2_fs W,b | 34,35 g2_gp
//  36-39 input_fs | 40-43 input_gp
//  44-49 out_head (W0,b0,W1,b1,W2,b2)
//  50,51 res_fs | 52,53 res_gp | 54-57 te_embed
//  58-61 x2_fs | 62-65 x2_gp | 66-69 x_post_fs | 70-73 x_post_gp
//  74-77 y2_fs | 78-81 y2_gp | 82-85 y_post_fs | 86-89 y_post_gp

extern "C" void kernel_launch(void* const* d_in, const int* in_sizes, int n_in,
                              void* d_out, int out_size, void* d_ws, size_t ws_size,
                              hipStream_t stream) {
    (void)in_sizes; (void)n_in; (void)out_size; (void)ws_size;
    auto F = [&](int i) { return (const float*)d_in[i]; };
    const int* TE = (const int*)d_in[4];

    char* ws = (char*)d_ws;
    const size_t NTDh  = (size_t)N_ * T_ * D_ * sizeof(_Float16);
    const size_t NTDf  = (size_t)N_ * T_ * D_ * sizeof(float);
    const size_t NTGh  = (size_t)N_ * T_ * 192 * sizeof(_Float16);
    _Float16* te      = (_Float16*)ws;                 size_t o = 4096;
    _Float16* xg2_gp  = (_Float16*)(ws + o); o += NTDh;
    _Float16* xg2_fs  = (_Float16*)(ws + o); o += NTDh;
    _Float16* u_gp    = (_Float16*)(ws + o); o += NTDh;
    _Float16* u_fs    = (_Float16*)(ws + o); o += NTDh;
    _Float16* gi_gp   = (_Float16*)(ws + o); o += NTGh;
    _Float16* gi_fs   = (_Float16*)(ws + o); o += NTGh;
    float*    pred_gp = (float*)(ws + o);    o += NTDf;
    float*    pred_fs = (float*)(ws + o);    o += NTDf;

    // ---- pre-pack all WMMA B operands (f32 -> f16 fragment-major) ----
    size_t po = o;
    auto pack = [&](int idx, int outd, int ind) -> const _Float16* {
        _Float16* dst = (_Float16*)(ws + po);
        po += ((size_t)outd * ind * sizeof(_Float16) + 255) & ~(size_t)255;
        k_pack<<<dim3(16), 256, 0, stream>>>(F(idx), dst, outd, ind);
        return dst;
    };

    // te embedding
    k_te<<<1, 256, 0, stream>>>(TE, F(54), F(55), F(56), F(57), te);

    // fused branch pipeline
    BranchParams Pg, Pf;
    Pg.x = F(0); Pg.w = F(2);
    Pg.in_W0 = F(40); Pg.in_b0 = F(41);
    Pg.in_W1p = pack(42, D_, D_); Pg.in_b1 = F(43);
    Pg.xp_W0p = pack(70, D_, D_); Pg.xp_b0 = F(71);
    Pg.xp_W1p = pack(72, D_, D_); Pg.xp_b1 = F(73);
    Pg.yp_W0p = pack(86, D_, D_); Pg.yp_b0 = F(87);
    Pg.yp_W1p = pack(88, D_, D_); Pg.yp_b1 = F(89);
    Pg.dl_W0p = pack(20, D_, D_); Pg.dl_b0 = F(21);
    Pg.dl_W1p = pack(22, D_, D_); Pg.dl_b1 = F(23);
    Pg.rs_Wp  = pack(52, D_, D_); Pg.rs_b  = F(53);
    Pg.x2_W0p = pack(62, D_, D_); Pg.x2_b0 = F(63);
    Pg.x2_W1p = pack(64, D_, D_); Pg.x2_b1 = F(65);
    Pg.y2_W0p = pack(78, D_, D_); Pg.y2_b0 = F(79);
    Pg.y2_W1p = pack(80, D_, D_); Pg.y2_b1 = F(81);
    Pg.g1_W0p = pack(28, D_, D_); Pg.g1_b0 = F(29);
    Pg.g1_W1p = pack(30, D_, D_); Pg.g1_b1 = F(31);
    Pg.xg2_out = xg2_gp; Pg.u_out = u_gp;

    Pf.x = F(1); Pf.w = F(3);
    Pf.in_W0 = F(36); Pf.in_b0 = F(37);
    Pf.in_W1p = pack(38, D_, D_); Pf.in_b1 = F(39);
    Pf.xp_W0p = pack(66, D_, D_); Pf.xp_b0 = F(67);
    Pf.xp_W1p = pack(68, D_, D_); Pf.xp_b1 = F(69);
    Pf.yp_W0p = pack(82, D_, D_); Pf.yp_b0 = F(83);
    Pf.yp_W1p = pack(84, D_, D_); Pf.yp_b1 = F(85);
    Pf.dl_W0p = pack(16, D_, D_); Pf.dl_b0 = F(17);
    Pf.dl_W1p = pack(18, D_, D_); Pf.dl_b1 = F(19);
    Pf.rs_Wp  = pack(50, D_, D_); Pf.rs_b  = F(51);
    Pf.x2_W0p = pack(58, D_, D_); Pf.x2_b0 = F(59);
    Pf.x2_W1p = pack(60, D_, D_); Pf.x2_b1 = F(61);
    Pf.y2_W0p = pack(74, D_, D_); Pf.y2_b0 = F(75);
    Pf.y2_W1p = pack(76, D_, D_); Pf.y2_b1 = F(77);
    Pf.g1_W0p = pack(24, D_, D_); Pf.g1_b0 = F(25);
    Pf.g1_W1p = pack(26, D_, D_); Pf.g1_b1 = F(27);
    Pf.xg2_out = xg2_fs; Pf.u_out = u_fs;

    size_t lds1 = (size_t)(2 * 256 * D_ + 4 * 16 * D_) * sizeof(_Float16) +
                  (256 + 128) * sizeof(float);  // ~75 KB < 320 KB/WGP
    k_branch<<<dim3(N_ / 16, T_, 2), 256, lds1, stream>>>(Pg, Pf);

    // GRU: parallel input projection + recurrent scan
    GruParams Gg, Gf;
    Gg.g2_Wp = pack(34, D_, D_);   Gg.g2_b = F(35);
    Gg.W_ihp = pack(13, 192, 128); Gg.b_ih = F(15);
    Gg.W_hhp = pack(12, 192, D_);  Gg.b_hh = F(14);
    Gg.xg2 = xg2_gp; Gg.u = u_gp; Gg.gi = gi_gp; Gg.pred = pred_gp;
    Gf.g2_Wp = pack(32, D_, D_);   Gf.g2_b = F(33);
    Gf.W_ihp = pack(9, 192, 128);  Gf.b_ih = F(11);
    Gf.W_hhp = pack(8, 192, D_);   Gf.b_hh = F(10);
    Gf.xg2 = xg2_fs; Gf.u = u_fs; Gf.gi = gi_fs; Gf.pred = pred_fs;
    k_gi<<<dim3(N_ / 16, T_, 2), 128, 0, stream>>>(Gg, Gf, te);
    k_gru<<<dim3(N_ / 16, 1, 2), 128, 0, stream>>>(Gg, Gf);

    // attention + output head
    HeadParams H;
    H.pred_gp = pred_gp; H.pred_fs = pred_fs;
    H.a1_Wp = pack(5, D_, D_); H.a1_b = F(6); H.a2_W = F(7);
    H.o0_Wp = pack(44, D_, D_); H.o0_b = F(45);
    H.o1_Wp = pack(46, D_, D_); H.o1_b = F(47);
    H.o2_W = F(48); H.o2_b = F(49);
    H.out = (float*)d_out;
    k_head<<<dim3((N_ * T_) / 16), 128, 0, stream>>>(H);
}
